// EfficientMDA_20323785244774
// MI455X (gfx1250) — matrix operand order repaired and checked by reading, not compiled, not run on hardware
//
#include <hip/hip_runtime.h>
#include <hip/hip_bf16.h>

typedef __attribute__((ext_vector_type(16))) _Float16     v16h;
typedef __attribute__((ext_vector_type(8)))  _Float16     v8h;
typedef __attribute__((ext_vector_type(8)))  float        v8f;
typedef __attribute__((ext_vector_type(8)))  unsigned int v8u;

#define B_  4
#define C_  256
#define H_  64
#define W_  64
#define R_  16
#define HW_ 4096

__device__ __forceinline__ float sigmoidf_(float v) {
    return 1.0f / (1.0f + __expf(-v));
}

// ---------------------------------------------------------------------------
// K0: one-time f32 -> f16 conversion of x (natural [b][c][i*64+j] order).
// ---------------------------------------------------------------------------
__global__ __launch_bounds__(256)
void k_half(const float* __restrict__ x, _Float16* __restrict__ xh16) {
    int t = blockIdx.x * blockDim.x + threadIdx.x;     // [0, 1048576)
    float4 f = ((const float4*)x)[t];
    v8h* dst4 = (v8h*)xh16;                            // write 4 halfs (8B)
    _Float16* d = xh16 + (size_t)t * 4;
    d[0] = (_Float16)f.x; d[1] = (_Float16)f.y;
    d[2] = (_Float16)f.z; d[3] = (_Float16)f.w;
    (void)dst4;
}

// ---------------------------------------------------------------------------
// K1: pool x over W (-> xh[b,c,h]) and over H (-> xw[b,c,w]).  131072 threads.
// ---------------------------------------------------------------------------
__global__ __launch_bounds__(256)
void k_pool(const float* __restrict__ x, float* __restrict__ xh,
            float* __restrict__ xw) {
    int t = blockIdx.x * blockDim.x + threadIdx.x;     // [0, 131072)
    int half = t >> 16;
    int idx  = t & 65535;                              // (b*C + c)*64 + p
    int bc   = idx >> 6;
    int p    = idx & 63;
    const float* base = x + (size_t)bc * HW_;
    float s = 0.0f;
    if (half == 0) {                                   // xh: sum over w (contig)
        const float* r = base + p * W_;
        #pragma unroll 8
        for (int w = 0; w < W_; ++w) s += r[w];
        xh[idx] = s * (1.0f / 64.0f);
    } else {                                           // xw: sum over h (stride)
        #pragma unroll 8
        for (int h = 0; h < H_; ++h) s += base[h * W_ + p];
        xw[idx] = s * (1.0f / 64.0f);
    }
}

// ---------------------------------------------------------------------------
// K2: h_vec[b,h,r] = w_reduce @ xh + b_reduce ; w_vec likewise.  8192 threads.
// ---------------------------------------------------------------------------
__global__ __launch_bounds__(256)
void k_vec(const float* __restrict__ xh, const float* __restrict__ xw,
           const float* __restrict__ w_reduce, const float* __restrict__ b_reduce,
           float* __restrict__ h_vec, float* __restrict__ w_vec) {
    int t = blockIdx.x * blockDim.x + threadIdx.x;     // [0, 8192)
    int half = (t >= B_ * 64 * R_) ? 1 : 0;
    int idx  = half ? t - B_ * 64 * R_ : t;            // (b*64 + p)*16 + r
    int r = idx & 15;
    int p = (idx >> 4) & 63;
    int b = idx >> 10;
    const float* src = half ? xw : xh;
    const float* wr  = w_reduce + r * C_;
    const float* sv  = src + (size_t)b * C_ * 64 + p;
    float s = b_reduce[r];
    #pragma unroll 8
    for (int c = 0; c < C_; ++c) s += wr[c] * sv[c * 64];
    (half ? w_vec : h_vec)[idx] = s;
}

// ---------------------------------------------------------------------------
// K3: SE chain: pooled -> hidden (relu) -> c_att (sigmoid).  One block.
// ---------------------------------------------------------------------------
__global__ __launch_bounds__(256)
void k_se(const float* __restrict__ h_vec,
          const float* __restrict__ w_c1, const float* __restrict__ b_c1,
          const float* __restrict__ w_c2, const float* __restrict__ b_c2,
          float* __restrict__ c_att) {
    __shared__ float pooled_s[B_ * R_];
    __shared__ float hidden_s[B_ * R_];
    int tid = threadIdx.x;
    if (tid < B_ * R_) {
        int b = tid >> 4, r = tid & 15;
        float s = 0.0f;
        for (int hh = 0; hh < H_; ++hh) s += h_vec[(b * 64 + hh) * R_ + r];
        pooled_s[tid] = s * (1.0f / 64.0f);
    }
    __syncthreads();
    if (tid < B_ * R_) {
        int b = tid >> 4, r = tid & 15;
        float s = b_c1[r];
        for (int r2 = 0; r2 < R_; ++r2) s += w_c1[r * R_ + r2] * pooled_s[b * R_ + r2];
        hidden_s[tid] = fmaxf(s, 0.0f);
    }
    __syncthreads();
    for (int k = 0; k < 4; ++k) {
        int idx = tid + k * 256;                       // b*C + c
        int b = idx >> 8, c = idx & 255;
        float s = b_c2[c];
        for (int r = 0; r < R_; ++r) s += w_c2[c * R_ + r] * hidden_s[b * R_ + r];
        c_att[idx] = sigmoidf_(s);
    }
}

// ---------------------------------------------------------------------------
// K4: h_att[b,h,i] and w_att[b,w,j] (sigmoid of R=16 dot).  32768 threads.
// ---------------------------------------------------------------------------
__global__ __launch_bounds__(256)
void k_att(const float* __restrict__ h_vec, const float* __restrict__ w_vec,
           const float* __restrict__ w_hfc, const float* __restrict__ b_hfc,
           const float* __restrict__ w_wfc, const float* __restrict__ b_wfc,
           float* __restrict__ h_att, float* __restrict__ w_att) {
    int t = blockIdx.x * blockDim.x + threadIdx.x;     // [0, 32768)
    int half = t >> 14;
    int idx  = t & 16383;                              // (b*64 + row)*64 + col
    int b   = idx >> 12;
    int row = (idx >> 6) & 63;
    int col = idx & 63;
    const float* vec = (half ? w_vec : h_vec) + (b * 64 + row) * R_;
    const float* wm  = (half ? w_wfc : w_hfc) + col * R_;
    float s = (half ? b_wfc : b_hfc)[col];
    #pragma unroll
    for (int r = 0; r < R_; ++r) s += vec[r] * wm[r];
    (half ? w_att : h_att)[idx] = sigmoidf_(s);
}

// ---------------------------------------------------------------------------
// K5: main factorized-softmax attention GEMM, f16 WMMA, f32 accumulate.
// Block = 512 threads (16 waves). Grid = (W/32, H, B).
// Wave m owns channels [16m, 16m+16). N-tile = 32 w's (two 16-wide accs).
// K = 4096 (= i*64 + j), staged 512 at a time: each wave builds one 32x32
// E-chunk (exp) into LDS; all waves consume all 16 chunks with WMMA.
// USE_F16X: A operand pre-converted to f16 in ws (no cvt in hot loop).
// ---------------------------------------------------------------------------
template <bool USE_F16X>
__global__ __launch_bounds__(512)
void k_main(const float* __restrict__ x, const _Float16* __restrict__ xh16,
            const float* __restrict__ h_att, const float* __restrict__ w_att,
            const float* __restrict__ c_att, const float* __restrict__ alpha,
            const float* __restrict__ beta, float* __restrict__ out) {
    __shared__ float        ah_s[64];           // h_att[b,h,:]
    __shared__ float        bw_s[32 * 64];      // w_att[b, wbase..+31, :]
    __shared__ unsigned int Bstage[16 * 2 * 32 * 8];  // 16 chunks x 2 subs, f16
    __shared__ float        den_s[32];

    const int b     = blockIdx.z;
    const int h     = blockIdx.y;
    const int wbase = blockIdx.x * 32;
    const int tid   = threadIdx.x;
    const int wave  = tid >> 5;
    const int lane  = tid & 31;
    const int nn    = lane & 15;
    const bool hi   = lane >= 16;

    if (tid < 64) ah_s[tid] = h_att[(b * 64 + h) * 64 + tid];
    for (int t = tid; t < 32 * 64; t += 512)
        bw_s[t] = w_att[(b * 64 + wbase + (t >> 6)) * 64 + (t & 63)];
    if (tid < 32) den_s[tid] = 0.0f;
    __syncthreads();

    const int cbase = wave * 16;
    const int ca    = cbase + nn;               // A-matrix row (M = lane%16)
    const float4*    xq  = (const float4*)(x + (size_t)(b * C_ + ca) * HW_);
    const _Float16*  xhp = xh16 + (size_t)(b * C_ + ca) * HW_;

    v8f acc0 = {}; v8f acc1 = {};
    float dsum0 = 0.0f, dsum1 = 0.0f;

    for (int s = 0; s < 8; ++s) {
        // Prefetch next stage's A region for this lane (global_prefetch_b8).
        if (s < 7) {
            if (USE_F16X) __builtin_prefetch(xhp + (s + 1) * 512 + wave * 32, 0, 0);
            else          __builtin_prefetch((const float*)xq + (s + 1) * 512 + wave * 32, 0, 0);
        }
        // ---- produce this wave's E chunk (32 K x 32 N) into LDS ----
        {
            const int   k0  = s * 512 + wave * 32;
            const int   i   = k0 >> 6;
            const int   j0  = (k0 & 63) + (hi ? 8 : 0);   // B layout: lo lanes K {0..7,16..23}, hi lanes K {8..15,24..31}
            const float ahi = ah_s[i];
            #pragma unroll
            for (int sub = 0; sub < 2; ++sub) {
                const float* bwrow = &bw_s[(sub * 16 + nn) * 64 + j0];
                v16h  bh;
                float ds = 0.0f;
                #pragma unroll
                for (int t = 0; t < 16; ++t) {
                    float e = __expf(ahi * bwrow[t + (t >= 8 ? 8 : 0)]);
                    ds += e;
                    bh[t] = (_Float16)e;
                }
                if (sub == 0) dsum0 += ds; else dsum1 += ds;
                *(v8u*)&Bstage[((wave * 2 + sub) * 32 + lane) * 8] =
                    __builtin_bit_cast(v8u, bh);
            }
        }
        __syncthreads();
        // ---- consume all 16 staged chunks ----
        #pragma unroll 4
        for (int ch = 0; ch < 16; ++ch) {
            const int k0 = s * 512 + ch * 32;
            // A-matrix lane layout: lo lanes K {0..7,16..23}, hi lanes K {8..15,24..31}
            v16h a;
            if (USE_F16X) {
                const _Float16* p = xhp + k0 + (hi ? 8 : 0);
                v8h alo = *(const v8h*)p;           // K j0 + {0..7}/{8..15}
                v8h ahi8 = *(const v8h*)(p + 16);   // K j0 + {16..23}/{24..31}
                a = __builtin_shufflevector(alo, ahi8,
                        0, 1, 2, 3, 4, 5, 6, 7, 8, 9, 10, 11, 12, 13, 14, 15);
            } else {
                const int i  = k0 >> 6;
                const int j0 = k0 & 63;
                const int q  = i * 16 + (j0 >> 2) + (hi ? 2 : 0);
                float4 f0 = xq[q];
                float4 f1 = xq[q + 1];
                float4 f2 = xq[q + 4];
                float4 f3 = xq[q + 5];
                a[0] = (_Float16)f0.x;  a[1] = (_Float16)f0.y;
                a[2] = (_Float16)f0.z;  a[3] = (_Float16)f0.w;
                a[4] = (_Float16)f1.x;  a[5] = (_Float16)f1.y;
                a[6] = (_Float16)f1.z;  a[7] = (_Float16)f1.w;
                a[8] = (_Float16)f2.x;  a[9] = (_Float16)f2.y;
                a[10] = (_Float16)f2.z; a[11] = (_Float16)f2.w;
                a[12] = (_Float16)f3.x; a[13] = (_Float16)f3.y;
                a[14] = (_Float16)f3.z; a[15] = (_Float16)f3.w;
            }
            v16h b0 = __builtin_bit_cast(v16h,
                          *(const v8u*)&Bstage[((ch * 2 + 0) * 32 + lane) * 8]);
            v16h b1 = __builtin_bit_cast(v16h,
                          *(const v8u*)&Bstage[((ch * 2 + 1) * 32 + lane) * 8]);
            acc0 = __builtin_amdgcn_wmma_f32_16x16x32_f16(
                       false, a, false, b0, (short)0, acc0, false, false);
            acc1 = __builtin_amdgcn_wmma_f32_16x16x32_f16(
                       false, a, false, b1, (short)0, acc1, false, false);
        }
        __syncthreads();
    }

    // ---- softmax denominators (each E value produced exactly once) ----
    atomicAdd(&den_s[nn], dsum0);
    atomicAdd(&den_s[16 + nn], dsum1);
    __syncthreads();

    // ---- epilogue: out = alpha * num/den + beta * x * c_att ----
    const float al   = alpha[0];
    const float bt   = beta[0];
    const float inv0 = 1.0f / den_s[nn];
    const float inv1 = 1.0f / den_s[16 + nn];
    const int   cofs = cbase + (hi ? 8 : 0);    // C/D layout: hi lanes are M+8
    #pragma unroll
    for (int v = 0; v < 8; ++v) {
        const int    c    = cofs + v;
        const float  cat  = c_att[b * C_ + c];
        const size_t rowo = (size_t)(b * C_ + c) * HW_ + (size_t)h * W_;
        {
            const int wg = wbase + nn;
            out[rowo + wg] = al * acc0[v] * inv0 + bt * x[rowo + wg] * cat;
        }
        {
            const int wg = wbase + 16 + nn;
            out[rowo + wg] = al * acc1[v] * inv1 + bt * x[rowo + wg] * cat;
        }
    }
}

// ---------------------------------------------------------------------------
extern "C" void kernel_launch(void* const* d_in, const int* in_sizes, int n_in,
                              void* d_out, int out_size, void* d_ws, size_t ws_size,
                              hipStream_t stream) {
    const float* x        = (const float*)d_in[0];
    const float* w_reduce = (const float*)d_in[1];
    const float* b_reduce = (const float*)d_in[2];
    const float* w_hfc    = (const float*)d_in[3];
    const float* b_hfc    = (const float*)d_in[4];
    const float* w_wfc    = (const float*)d_in[5];
    const float* b_wfc    = (const float*)d_in[6];
    const float* w_c1     = (const float*)d_in[7];
    const float* b_c1     = (const float*)d_in[8];
    const float* w_c2     = (const float*)d_in[9];
    const float* b_c2     = (const float*)d_in[10];
    const float* alpha    = (const float*)d_in[11];
    const float* beta     = (const float*)d_in[12];
    float*       out      = (float*)d_out;

    const size_t N_ELEM    = (size_t)B_ * C_ * HW_;          // 4,194,304
    const size_t F16_BYTES = N_ELEM * sizeof(_Float16);      // 8 MB
    const size_t SMALL_FLOATS = 173056;                      // helper arrays
    const bool   use_f16x = ws_size >= F16_BYTES + SMALL_FLOATS * sizeof(float);

    _Float16* xh16 = (_Float16*)d_ws;                        // 8 MB (f16 path)
    float* ws    = use_f16x ? (float*)((char*)d_ws + F16_BYTES) : (float*)d_ws;
    float* xh    = ws;                 // 65536
    float* xw    = ws + 65536;         // 65536
    float* h_vec = ws + 131072;        // 4096
    float* w_vec = ws + 135168;        // 4096
    float* c_att = ws + 139264;        // 1024
    float* h_att = ws + 140288;        // 16384
    float* w_att = ws + 156672;        // 16384   (total 173056 floats ~ 676 KB)

    if (use_f16x)
        k_half<<<4096, 256, 0, stream>>>(x, xh16);
    k_pool<<<512, 256, 0, stream>>>(x, xh, xw);
    k_vec <<<32,  256, 0, stream>>>(xh, xw, w_reduce, b_reduce, h_vec, w_vec);
    k_se  <<<1,   256, 0, stream>>>(h_vec, w_c1, b_c1, w_c2, b_c2, c_att);
    k_att <<<128, 256, 0, stream>>>(h_vec, w_vec, w_hfc, b_hfc, w_wfc, b_wfc,
                                    h_att, w_att);
    if (use_f16x)
        k_main<true><<<dim3(2, 64, 4), 512, 0, stream>>>(
            x, xh16, h_att, w_att, c_att, alpha, beta, out);
    else
        k_main<false><<<dim3(2, 64, 4), 512, 0, stream>>>(
            x, xh16, h_att, w_att, c_att, alpha, beta, out);
}
